// GAT_GCN_2963527434725
// MI455X (gfx1250) — matrix operand order repaired and checked by reading, not compiled, not run on hardware
//
#include <hip/hip_runtime.h>
#include <hip/hip_bf16.h>

#define NN 16384              // nodes
#define EE 131072             // edges (before self loops)
#define ET (EE + NN)          // edges + self loops = 147456
#define GG 128                // graphs
#define FF 78                 // feature dim
#define HH 10                 // GAT heads
#define HFD 780               // H*F
#define CL 594                // conv output length (625 - 32 + 1)

typedef __attribute__((ext_vector_type(16))) _Float16 v16h;
typedef __attribute__((ext_vector_type(8)))  _Float16 v8h;
typedef __attribute__((ext_vector_type(8)))  float    v8f;

static __device__ __forceinline__ void atomicMaxF(float* addr, float val) {
  int* ia = (int*)addr;
  int old = *ia;
  while (__int_as_float(old) < val) {
    int assumed = old;
    old = atomicCAS(ia, assumed, __float_as_int(val));
    if (old == assumed) break;
  }
}

// ======================================================================
// Operand packing: f32 -> padded f16.  A stays row-major [M,Kpad];
// B is packed transposed [Npad,Kpad] so each lane's K-run is contiguous.
// ======================================================================
__global__ void pack_a_f16(const float* __restrict__ src, _Float16* __restrict__ dst,
                           int M, int K, int Kpad) {
  int i = blockIdx.x * blockDim.x + threadIdx.x;
  if (i >= M * Kpad) return;
  int m = i / Kpad, k = i % Kpad;
  dst[i] = (_Float16)(k < K ? src[(size_t)m * K + k] : 0.f);
}

__global__ void pack_bt_f16(const float* __restrict__ src, _Float16* __restrict__ dst,
                            int K, int N, int Kpad, int Npad) {
  int i = blockIdx.x * blockDim.x + threadIdx.x;
  if (i >= Npad * Kpad) return;
  int n = i / Kpad, k = i % Kpad;
  dst[i] = (_Float16)((k < K && n < N) ? src[(size_t)k * N + n] : 0.f);
}

// ======================================================================
// WMMA GEMM on packed f16 operands: C[M,N] = A @ B (+bias) (opt ReLU).
// One wave32 per 32x32 output tile: 2 A frags x 2 B frags -> 4 WMMAs
// per 32-K step, 8x global_load_b128 (2 loads per WMMA), branch-free.
// A/B K order: contiguous, k = 16*half + s in both operands.
// ======================================================================
__global__ __launch_bounds__(32)
void wmma_gemm_f16(const _Float16* __restrict__ A, const _Float16* __restrict__ Bt,
                   const float* __restrict__ bias, float* __restrict__ C,
                   int M, int N, int Kpad, int ldc, int relu) {
  const int lane   = threadIdx.x & 31;
  const int half16 = lane >> 4;          // K-half selector
  const int lm     = lane & 15;
  const int tM = blockIdx.y * 32;
  const int tN = blockIdx.x * 32;

  const _Float16* ap0 = A  + (size_t)(tM + lm) * Kpad;        // rows tM..tM+15
  const _Float16* ap1 = A  + (size_t)(tM + 16 + lm) * Kpad;   // rows tM+16..tM+31
  const _Float16* bp0 = Bt + (size_t)(tN + lm) * Kpad      + 16 * half16;
  const _Float16* bp1 = Bt + (size_t)(tN + 16 + lm) * Kpad + 16 * half16;

  v8f acc00 = {}, acc01 = {}, acc10 = {}, acc11 = {};
  for (int k0 = 0; k0 < Kpad; k0 += 32) {
    // A 16x32 fragments: lane holds row; two contiguous 16B chunks
    // (K = 0..15 for half 0, 16..31 for half 1)
    v8h a0lo = *(const v8h*)(ap0 + k0 + 16 * half16);
    v8h a0hi = *(const v8h*)(ap0 + k0 + 8 + 16 * half16);
    v8h a1lo = *(const v8h*)(ap1 + k0 + 16 * half16);
    v8h a1hi = *(const v8h*)(ap1 + k0 + 8 + 16 * half16);
    v16h a0 = __builtin_shufflevector(a0lo, a0hi,
                                      0, 1, 2, 3, 4, 5, 6, 7,
                                      8, 9, 10, 11, 12, 13, 14, 15);
    v16h a1 = __builtin_shufflevector(a1lo, a1hi,
                                      0, 1, 2, 3, 4, 5, 6, 7,
                                      8, 9, 10, 11, 12, 13, 14, 15);
    // B 32x16 fragments: lane holds column; contiguous 32B K-run
    v16h b0 = *(const v16h*)(bp0 + k0);
    v16h b1 = *(const v16h*)(bp1 + k0);
    acc00 = __builtin_amdgcn_wmma_f32_16x16x32_f16(false, a0, false, b0,
                                                   (short)0, acc00, false, false);
    acc01 = __builtin_amdgcn_wmma_f32_16x16x32_f16(false, a0, false, b1,
                                                   (short)0, acc01, false, false);
    acc10 = __builtin_amdgcn_wmma_f32_16x16x32_f16(false, a1, false, b0,
                                                   (short)0, acc10, false, false);
    acc11 = __builtin_amdgcn_wmma_f32_16x16x32_f16(false, a1, false, b1,
                                                   (short)0, acc11, false, false);
  }
  const int c0 = tN + lm, c1 = tN + 16 + lm;
#pragma unroll
  for (int r = 0; r < 8; ++r) {          // C/D: lane 0-15 M=r, lane 16-31 M=8+r
    int r0 = tM + r + 8 * half16;        // M is always a multiple of 32
    int r1 = r0 + 16;
    if (c0 < N) {
      float v = acc00[r];
      if (bias) v += bias[c0];
      if (relu) v = fmaxf(v, 0.f);
      C[(size_t)r0 * ldc + c0] = v;
      float w = acc10[r];
      if (bias) w += bias[c0];
      if (relu) w = fmaxf(w, 0.f);
      C[(size_t)r1 * ldc + c0] = w;
    }
    if (c1 < N) {
      float v = acc01[r];
      if (bias) v += bias[c1];
      if (relu) v = fmaxf(v, 0.f);
      C[(size_t)r0 * ldc + c1] = v;
      float w = acc11[r];
      if (bias) w += bias[c1];
      if (relu) w = fmaxf(w, 0.f);
      C[(size_t)r1 * ldc + c1] = w;
    }
  }
}

// ======================================================================
// Elementwise / scatter helpers
// ======================================================================
__global__ void fill_kernel(float* __restrict__ p, float v, int n) {
  int i = blockIdx.x * blockDim.x + threadIdx.x;
  if (i < n) p[i] = v;
}

__global__ void rowbias_kernel(float* __restrict__ out, const float* __restrict__ bias,
                               int n, int D) {
  int i = blockIdx.x * blockDim.x + threadIdx.x;
  if (i < n) out[i] = bias[i % D];
}

__global__ void relu_kernel(float* __restrict__ p, int n) {
  int i = blockIdx.x * blockDim.x + threadIdx.x;
  if (i < n) p[i] = fmaxf(p[i], 0.f);
}

__global__ void fix_nonfinite(float* __restrict__ p, int n) {
  int i = blockIdx.x * blockDim.x + threadIdx.x;
  if (i >= n) return;
  float v = p[i];
  if (!(v > -__builtin_inff() && v < __builtin_inff())) p[i] = 0.f;
}

__global__ void count_kernel(const int* __restrict__ batch, float* __restrict__ cnt) {
  int n = blockIdx.x * blockDim.x + threadIdx.x;
  if (n < NN) atomicAdd(&cnt[batch[n]], 1.0f);
}

// ======================================================================
// GAT stages
// ======================================================================
__global__ void gat_scores(const float* __restrict__ h, const float* __restrict__ a_s,
                           const float* __restrict__ a_d,
                           float* __restrict__ asum, float* __restrict__ adsum) {
  int idx = blockIdx.x * blockDim.x + threadIdx.x;
  if (idx >= NN * HH) return;
  int n = idx / HH, hh = idx % HH;
  const float* hp = h + (size_t)n * HFD + hh * FF;
  const float* ap = a_s + hh * FF;
  const float* dp = a_d + hh * FF;
  float s1 = 0.f, s2 = 0.f;
  for (int f = 0; f < FF; ++f) { float v = hp[f]; s1 += v * ap[f]; s2 += v * dp[f]; }
  asum[idx] = s1; adsum[idx] = s2;
}

__global__ void gat_edge_max(const int* __restrict__ ei, const float* __restrict__ asum,
                             const float* __restrict__ adsum,
                             float* __restrict__ e, float* __restrict__ m) {
  int t = blockIdx.x * blockDim.x + threadIdx.x;
  if (t >= ET * HH) return;
  int edge = t / HH, hh = t % HH;
  int src = edge < EE ? ei[edge] : edge - EE;
  int dst = edge < EE ? ei[EE + edge] : edge - EE;
  float v = asum[src * HH + hh] + adsum[dst * HH + hh];
  v = v > 0.f ? v : 0.2f * v;
  e[t] = v;
  atomicMaxF(&m[dst * HH + hh], v);
}

__global__ void gat_edge_exp(const int* __restrict__ ei, float* __restrict__ e,
                             const float* __restrict__ m, float* __restrict__ s) {
  int t = blockIdx.x * blockDim.x + threadIdx.x;
  if (t >= ET * HH) return;
  int edge = t / HH, hh = t % HH;
  int dst = edge < EE ? ei[EE + edge] : edge - EE;
  float ex = __expf(e[t] - m[dst * HH + hh]);
  e[t] = ex;
  atomicAdd(&s[dst * HH + hh], ex);
}

__global__ void gat_aggregate(const int* __restrict__ ei, const float* __restrict__ h,
                              const float* __restrict__ ex, const float* __restrict__ s,
                              float* __restrict__ out) {
  __shared__ float al[HH];
  int edge = blockIdx.x;
  int src = edge < EE ? ei[edge] : edge - EE;
  int dst = edge < EE ? ei[EE + edge] : edge - EE;
  if (threadIdx.x < HH)
    al[threadIdx.x] = ex[(size_t)edge * HH + threadIdx.x] / s[dst * HH + threadIdx.x];
  __syncthreads();
  const float* hp = h + (size_t)src * HFD;
  float* op = out + (size_t)dst * HFD;
  for (int j = threadIdx.x; j < HFD; j += blockDim.x)
    atomicAdd(&op[j], al[j / FF] * hp[j]);
}

// ======================================================================
// GCN stages
// ======================================================================
__global__ void gcn_deg(const int* __restrict__ ei, float* __restrict__ deg) {
  int t = blockIdx.x * blockDim.x + threadIdx.x;
  if (t >= ET) return;
  int dst = t < EE ? ei[EE + t] : t - EE;
  atomicAdd(&deg[dst], 1.0f);
}

__global__ void gcn_dinv(const float* __restrict__ deg, float* __restrict__ dinv) {
  int i = blockIdx.x * blockDim.x + threadIdx.x;
  if (i >= NN) return;
  float d = deg[i];
  dinv[i] = d > 0.f ? rsqrtf(fmaxf(d, 1e-12f)) : 0.f;
}

__global__ void gcn_agg(const int* __restrict__ ei, const float* __restrict__ h,
                        const float* __restrict__ dinv, float* __restrict__ out) {
  int t = blockIdx.x * blockDim.x + threadIdx.x;
  if (t >= ET * FF) return;
  int edge = t / FF, f = t % FF;
  int src = edge < EE ? ei[edge] : edge - EE;
  int dst = edge < EE ? ei[EE + edge] : edge - EE;
  float nrm = dinv[src] * dinv[dst];
  atomicAdd(&out[(size_t)dst * FF + f], nrm * h[(size_t)src * FF + f]);
}

// ======================================================================
// Pooling: pool[g, 0:D] = segment max, pool[g, D:2D] = segment mean
// ======================================================================
__global__ void pool_init(float* __restrict__ pool, int D) {
  int t = blockIdx.x * blockDim.x + threadIdx.x;
  if (t >= GG * 2 * D) return;
  pool[t] = (t % (2 * D)) < D ? -__builtin_inff() : 0.f;
}

__global__ void pool_scatter(const float* __restrict__ v, const int* __restrict__ batch,
                             float* __restrict__ pool, int D) {
  int t = blockIdx.x * blockDim.x + threadIdx.x;
  if (t >= NN * D) return;
  int n = t / D, d = t % D;
  int g = batch[n];
  float x = v[t];
  atomicMaxF(&pool[(size_t)g * 2 * D + d], x);
  atomicAdd(&pool[(size_t)g * 2 * D + D + d], x);
}

__global__ void pool_finalize(float* __restrict__ pool, const float* __restrict__ cnt, int D) {
  int t = blockIdx.x * blockDim.x + threadIdx.x;
  if (t >= GG * D) return;
  int g = t / D, d = t % D;
  float mx = pool[(size_t)g * 2 * D + d];
  if (!(mx > -__builtin_inff() && mx < __builtin_inff())) mx = 0.f;
  pool[(size_t)g * 2 * D + d] = mx;
  pool[(size_t)g * 2 * D + D + d] /= fmaxf(cnt[g], 1.0f);
}

// ======================================================================
// Protein branch: Conv1d(1,1,32) valid + ReLU -> [G, 594]
// ======================================================================
__global__ void conv1d_kernel(const float* __restrict__ target, const float* __restrict__ w,
                              const float* __restrict__ b, float* __restrict__ y) {
  int t = blockIdx.x * blockDim.x + threadIdx.x;
  if (t >= GG * CL) return;
  int g = t / CL, tt = t % CL;
  const float* tp = target + (size_t)g * 625 + tt;
  float s = b[0];
#pragma unroll
  for (int k = 0; k < 32; ++k) s += tp[k] * w[k];
  y[t] = fmaxf(s, 0.f);
}

// ======================================================================
extern "C" void kernel_launch(void* const* d_in, const int* in_sizes, int n_in,
                              void* d_out, int out_size, void* d_ws, size_t ws_size,
                              hipStream_t stream) {
  (void)in_sizes; (void)n_in; (void)out_size; (void)ws_size;
  const float* x      = (const float*)d_in[0];
  const int*   ei     = (const int*)d_in[1];
  const int*   batch  = (const int*)d_in[2];
  const float* target = (const float*)d_in[3];
  const float* Wg1  = (const float*)d_in[4];
  const float* as1  = (const float*)d_in[5];
  const float* ad1  = (const float*)d_in[6];
  const float* bg1  = (const float*)d_in[7];
  const float* Wg2  = (const float*)d_in[8];
  const float* as2  = (const float*)d_in[9];
  const float* ad2  = (const float*)d_in[10];
  const float* bg2  = (const float*)d_in[11];
  const float* Wgcn = (const float*)d_in[12];
  const float* bgcn = (const float*)d_in[13];
  const float* Wfg1 = (const float*)d_in[14];
  const float* bfg1 = (const float*)d_in[15];
  const float* Wfg2 = (const float*)d_in[16];
  const float* bfg2 = (const float*)d_in[17];
  const float* wconv= (const float*)d_in[18];
  const float* bconv= (const float*)d_in[19];
  const float* Wxt  = (const float*)d_in[20];
  const float* bxt  = (const float*)d_in[21];
  const float* W1   = (const float*)d_in[22];
  const float* b1   = (const float*)d_in[23];
  const float* W2   = (const float*)d_in[24];
  const float* b2   = (const float*)d_in[25];
  const float* Wo   = (const float*)d_in[26];
  const float* bo   = (const float*)d_in[27];
  float* out = (float*)d_out;

  // ---- workspace carve-out (256B aligned) ----
  char* ws = (char*)d_ws;
  size_t off = 0;
  auto alloc = [&](size_t bytes) -> void* {
    void* p = (void*)(ws + off);
    off += (bytes + 255) & ~(size_t)255;
    return p;
  };
  float* bufA  = (float*)alloc((size_t)NN * HFD * 4);   // h1, then h2
  float* bufB  = (float*)alloc((size_t)NN * HFD * 4);   // x1, then GAT2 output
  float* asum  = (float*)alloc((size_t)NN * HH * 4);
  float* adsum = (float*)alloc((size_t)NN * HH * 4);
  float* mbuf  = (float*)alloc((size_t)NN * HH * 4);
  float* sbuf  = (float*)alloc((size_t)NN * HH * 4);
  float* ebuf  = (float*)alloc((size_t)ET * HH * 4);
  float* hg    = (float*)alloc((size_t)NN * FF * 4);    // GCN h (both passes)
  float* ga1   = (float*)alloc((size_t)NN * FF * 4);
  float* ga2   = (float*)alloc((size_t)NN * FF * 4);
  float* deg   = (float*)alloc(NN * 4);
  float* dinv  = (float*)alloc(NN * 4);
  float* cnt   = (float*)alloc(GG * 4);
  float* pool1 = (float*)alloc((size_t)GG * 2 * HFD * 4);
  float* pool2 = (float*)alloc((size_t)GG * 2 * FF * 4);
  float* convy = (float*)alloc((size_t)GG * CL * 4);
  float* xc    = (float*)alloc((size_t)GG * 512 * 4);
  float* f1    = (float*)alloc((size_t)GG * 512 * 4);
  float* f2    = (float*)alloc((size_t)GG * 256 * 4);
  _Float16* bufPA = (_Float16*)alloc((size_t)NN * 800 * 2);   // packed A (max 16384x800)
  _Float16* bufPB = (_Float16*)alloc((size_t)800 * 1600 * 2); // packed B^T (max Npad*Kpad)

  auto cdiv = [](int a, int b) { return (a + b - 1) / b; };
  const int T = 256;
  const float NEGINF = -__builtin_inff();

  // packed-f16 WMMA GEMM wrapper: pack A row-major, B transposed, then GEMM
  // (every M here is a multiple of 32: 16384 node rows or 128 graph rows)
  auto gemm = [&](const float* Af, const float* Bf, const float* bias, float* Cc,
                  int M, int N, int K, int ldc, int relu) {
    int Kpad = (K + 31) & ~31;
    int Npad = (N + 31) & ~31;
    pack_a_f16<<<cdiv(M * Kpad, T), T, 0, stream>>>(Af, bufPA, M, K, Kpad);
    pack_bt_f16<<<cdiv(Npad * Kpad, T), T, 0, stream>>>(Bf, bufPB, K, N, Kpad, Npad);
    wmma_gemm_f16<<<dim3(Npad / 32, M / 32), 32, 0, stream>>>(
        bufPA, bufPB, bias, Cc, M, N, Kpad, ldc, relu);
  };

  // nodes-per-graph counts (shared by both pools)
  fill_kernel<<<cdiv(GG, T), T, 0, stream>>>(cnt, 0.f, GG);
  count_kernel<<<cdiv(NN, T), T, 0, stream>>>(batch, cnt);

  // ===== GAT layer 1: h1 = x @ Wg1 =====
  gemm(x, Wg1, nullptr, bufA, NN, HFD, FF, HFD, 0);
  gat_scores<<<cdiv(NN * HH, T), T, 0, stream>>>(bufA, as1, ad1, asum, adsum);
  fill_kernel<<<cdiv(NN * HH, T), T, 0, stream>>>(mbuf, NEGINF, NN * HH);
  fill_kernel<<<cdiv(NN * HH, T), T, 0, stream>>>(sbuf, 0.f, NN * HH);
  gat_edge_max<<<cdiv(ET * HH, T), T, 0, stream>>>(ei, asum, adsum, ebuf, mbuf);
  fix_nonfinite<<<cdiv(NN * HH, T), T, 0, stream>>>(mbuf, NN * HH);
  gat_edge_exp<<<cdiv(ET * HH, T), T, 0, stream>>>(ei, ebuf, mbuf, sbuf);
  rowbias_kernel<<<cdiv(NN * HFD, T), T, 0, stream>>>(bufB, bg1, NN * HFD, HFD);
  gat_aggregate<<<ET, T, 0, stream>>>(ei, bufA, ebuf, sbuf, bufB);   // bufB = x1

  // ===== GAT layer 2: h2 = x1 @ Wg2 (the 20 GFLOP GEMM) =====
  gemm(bufB, Wg2, nullptr, bufA, NN, HFD, HFD, HFD, 0);
  gat_scores<<<cdiv(NN * HH, T), T, 0, stream>>>(bufA, as2, ad2, asum, adsum);
  fill_kernel<<<cdiv(NN * HH, T), T, 0, stream>>>(mbuf, NEGINF, NN * HH);
  fill_kernel<<<cdiv(NN * HH, T), T, 0, stream>>>(sbuf, 0.f, NN * HH);
  gat_edge_max<<<cdiv(ET * HH, T), T, 0, stream>>>(ei, asum, adsum, ebuf, mbuf);
  fix_nonfinite<<<cdiv(NN * HH, T), T, 0, stream>>>(mbuf, NN * HH);
  gat_edge_exp<<<cdiv(ET * HH, T), T, 0, stream>>>(ei, ebuf, mbuf, sbuf);
  rowbias_kernel<<<cdiv(NN * HFD, T), T, 0, stream>>>(bufB, bg2, NN * HFD, HFD);
  gat_aggregate<<<ET, T, 0, stream>>>(ei, bufA, ebuf, sbuf, bufB);
  relu_kernel<<<cdiv(NN * HFD, T), T, 0, stream>>>(bufB, NN * HFD);

  pool_init<<<cdiv(GG * 2 * HFD, T), T, 0, stream>>>(pool1, HFD);
  pool_scatter<<<cdiv(NN * HFD, T), T, 0, stream>>>(bufB, batch, pool1, HFD);
  pool_finalize<<<cdiv(GG * HFD, T), T, 0, stream>>>(pool1, cnt, HFD);

  // ===== GCN branch (same W applied twice) =====
  fill_kernel<<<cdiv(NN, T), T, 0, stream>>>(deg, 0.f, NN);
  gcn_deg<<<cdiv(ET, T), T, 0, stream>>>(ei, deg);
  gcn_dinv<<<cdiv(NN, T), T, 0, stream>>>(deg, dinv);

  gemm(x, Wgcn, nullptr, hg, NN, FF, FF, FF, 0);
  rowbias_kernel<<<cdiv(NN * FF, T), T, 0, stream>>>(ga1, bgcn, NN * FF, FF);
  gcn_agg<<<cdiv(ET * FF, T), T, 0, stream>>>(ei, hg, dinv, ga1);

  gemm(ga1, Wgcn, nullptr, hg, NN, FF, FF, FF, 0);
  rowbias_kernel<<<cdiv(NN * FF, T), T, 0, stream>>>(ga2, bgcn, NN * FF, FF);
  gcn_agg<<<cdiv(ET * FF, T), T, 0, stream>>>(ei, hg, dinv, ga2);
  relu_kernel<<<cdiv(NN * FF, T), T, 0, stream>>>(ga2, NN * FF);

  pool_init<<<cdiv(GG * 2 * FF, T), T, 0, stream>>>(pool2, FF);
  pool_scatter<<<cdiv(NN * FF, T), T, 0, stream>>>(ga2, batch, pool2, FF);
  pool_finalize<<<cdiv(GG * FF, T), T, 0, stream>>>(pool2, cnt, FF);

  // ===== protein branch =====
  conv1d_kernel<<<cdiv(GG * CL, T), T, 0, stream>>>(target, wconv, bconv, convy);

  // ===== heads: write directly into xc = [fg1 | fg2 | xt] (ldc = 512) =====
  gemm(pool1, Wfg1, bfg1, xc + 0,   GG, 128, 2 * HFD, 512, 1);
  gemm(pool2, Wfg2, bfg2, xc + 128, GG, 128, 2 * FF,  512, 1);
  gemm(convy, Wxt,  bxt,  xc + 256, GG, 256, CL,      512, 0);

  // ===== MLP head =====
  gemm(xc, W1, b1, f1, GG, 512, 512, 512, 1);
  gemm(f1, W2, b2, f2, GG, 256, 512, 256, 1);
  gemm(f2, Wo, bo, out, GG, 1, 256, 1, 0);
}
